// MultiHead_11622181503171
// MI455X (gfx1250) — compile-verified, matrix operand order checked by reading, not run on hardware
//
#include <hip/hip_runtime.h>
#include <hip/hip_bf16.h>

// ---------------- problem constants ----------------
constexpr int Bc = 2;
constexpr int Tc = 2048;
constexpr int Cc = 1024;   // n_embd
constexpr int Hc = 16;     // heads
constexpr int Dc = 64;     // head size
constexpr int KD = Hc * Dc;  // 1024

typedef __attribute__((ext_vector_type(16))) _Float16 v16h;
typedef __attribute__((ext_vector_type(8)))  _Float16 v8h;
typedef __attribute__((ext_vector_type(8)))  float    v8f;
typedef unsigned int u32x4 __attribute__((ext_vector_type(4)));
typedef int          i32x4 __attribute__((ext_vector_type(4)));
typedef int          i32x8 __attribute__((ext_vector_type(8)));

#if defined(__has_builtin)
#if __has_builtin(__builtin_amdgcn_tensor_load_to_lds)
#define HAS_TDM 1
#else
#define HAS_TDM 0
#endif
#else
#define HAS_TDM 0
#endif

__device__ __forceinline__ v8f zero8() {
    v8f z = {0.f, 0.f, 0.f, 0.f, 0.f, 0.f, 0.f, 0.f};
    return z;
}

// WMMA f32 = f16 x f16 + f32, 16x16x32
__device__ __forceinline__ v8f wmma32(v16h a, v16h b, v8f c) {
    return __builtin_amdgcn_wmma_f32_16x16x32_f16(
        /*neg_a=*/false, a, /*neg_b=*/false, b,
        /*c_mod=*/(short)0, c, /*reuse_a=*/false, /*reuse_b=*/false);
}

// Load a 16x32 f16 A/B fragment (ISA 16-bit layout): lane l -> idx = l&15,
// lanes 0-15: K=[0..7],[16..23]; lanes 16-31: K=[8..15],[24..31]
// => two contiguous 16B loads at kb and kb+16, kb = (l>>4)*8.
__device__ __forceinline__ v16h load_frag(const _Float16* __restrict__ p, int ld) {
    const int lane = threadIdx.x & 31;
    const int idx  = lane & 15;
    const int kb   = (lane >> 4) << 3;
    const _Float16* r = p + (size_t)idx * (size_t)ld + kb;
    union { v16h v; v8h h[2]; } u;
    u.h[0] = *(const v8h*)(r);
    u.h[1] = *(const v8h*)(r + 16);
    return u.v;
}

__device__ __forceinline__ v8f vmax8(v8f a, v8f b) {
    v8f r;
#pragma unroll
    for (int i = 0; i < 8; ++i) r[i] = fmaxf(a[i], b[i]);
    return r;
}
__device__ __forceinline__ v8f reduce16_max(v8f v) {
#pragma unroll
    for (int m = 1; m < 16; m <<= 1) {
        v8f o;
#pragma unroll
        for (int i = 0; i < 8; ++i) o[i] = __shfl_xor(v[i], m, 32);
        v = vmax8(v, o);
    }
    return v;
}
__device__ __forceinline__ v8f reduce16_sum(v8f v) {
#pragma unroll
    for (int m = 1; m < 16; m <<= 1) {
#pragma unroll
        for (int i = 0; i < 8; ++i) v[i] += __shfl_xor(v[i], m, 32);
    }
    return v;
}

#if HAS_TDM
// Tensor Data Mover: DMA a (tile_d1 rows x tile_d0 elems) f16 tile from global
// memory (row stride = row_stride_elems) into LDS at lds_byte_addr.
// D# group0: count=1 | lds_addr | global_addr[56:0] | type=2
// D# group1: data_size=1 (2B), tensor_dim0/1 = tile dims, dim0 stride.
__device__ __forceinline__ void tdm_load_tile_f16(
    const _Float16* gsrc, unsigned int lds_byte_addr,
    unsigned int tile_d0, unsigned int tile_d1,
    unsigned long long row_stride_elems) {
    unsigned long long ga = (unsigned long long)(uintptr_t)gsrc;
    u32x4 g0 = { 1u,                                    // count=1
                 lds_byte_addr,                         // lds_addr
                 (unsigned int)ga,                      // global_addr[31:0]
                 (unsigned int)((ga >> 32) & 0x1FFFFFFu) | (2u << 30) }; // [56:32]|type=2
    i32x8 g1;
    g1[0] = (int)(1u << 16);                            // data_size=1 -> 2 bytes
    g1[1] = (int)((tile_d0 & 0xFFFFu) << 16);           // tensor_dim0[15:0]
    g1[2] = (int)((tile_d1 & 0xFFFFu) << 16);           // tensor_dim0 hi=0 | tensor_dim1 lo
    g1[3] = (int)((tile_d0 & 0xFFFFu) << 16);           // tensor_dim1 hi=0 | tile_dim0
    g1[4] = (int)(tile_d1 & 0xFFFFu);                   // tile_dim1 | tile_dim2=0
    g1[5] = (int)(unsigned int)(row_stride_elems & 0xFFFFFFFFu);  // dim0_stride[31:0]
    g1[6] = (int)(unsigned int)((row_stride_elems >> 32) & 0xFFFFu); // stride hi | dim1_stride lo=0
    g1[7] = 0;
    i32x4 z4 = {0, 0, 0, 0};
#if __clang_major__ >= 23
    i32x8 z8 = {0, 0, 0, 0, 0, 0, 0, 0};
    __builtin_amdgcn_tensor_load_to_lds(g0, g1, z4, z4, z8, 0);
#else
    __builtin_amdgcn_tensor_load_to_lds(g0, g1, z4, z4, 0);
#endif
}
#endif

// ---------------- conversion / transpose kernels ----------------
__global__ void k_cvt(const float* __restrict__ s, _Float16* __restrict__ d, int n) {
    int i = blockIdx.x * blockDim.x + threadIdx.x;
    if (i < n) d[i] = (_Float16)s[i];
}

// W [H,C,D] f32 -> WT [H,D,C] f16
__global__ void k_cvt_tr_w(const float* __restrict__ W, _Float16* __restrict__ WT) {
    int i = blockIdx.x * blockDim.x + threadIdx.x;
    if (i >= Hc * Dc * Cc) return;
    int c = i % Cc;
    int t = i / Cc;
    int d = t % Dc;
    int h = t / Dc;
    WT[i] = (_Float16)W[((size_t)h * Cc + c) * Dc + d];
}

// Wo [KD,C] f32 -> WoT [C,KD] f16
__global__ void k_cvt_tr_wo(const float* __restrict__ W, _Float16* __restrict__ WT) {
    int i = blockIdx.x * blockDim.x + threadIdx.x;
    if (i >= Cc * KD) return;
    int k = i % KD;
    int c = i / KD;
    WT[i] = (_Float16)W[(size_t)k * Cc + c];
}

// ---------------- QKV projection ----------------
// grid: B*H*(T/64) blocks, 128 threads (4 waves); wave owns 16 rows x D=64 for q,k,v
__global__ __launch_bounds__(128) void k_qkv(
    const _Float16* __restrict__ xh,                       // [B,T,C]
    const _Float16* __restrict__ wqT,                      // [H,D,C]
    const _Float16* __restrict__ wkT,
    const _Float16* __restrict__ wvT,
    _Float16* __restrict__ qh,                             // [B,H,T,D]
    _Float16* __restrict__ kh,                             // [B,H,T,D]
    _Float16* __restrict__ vT) {                           // [B,H,D,T]
    const int ntb  = Tc / 64;
    const int tb   = blockIdx.x % ntb;
    const int bh   = blockIdx.x / ntb;
    const int h    = bh % Hc;
    const int b    = bh / Hc;
    const int wave = threadIdx.x >> 5;
    const int lane = threadIdx.x & 31;
    const int ln   = lane & 15;
    const int half = lane >> 4;
    const int row0 = tb * 64 + wave * 16;

    const _Float16* A  = xh  + ((size_t)b * Tc + row0) * Cc;
    const _Float16* BQ = wqT + (size_t)h * Dc * Cc;
    const _Float16* BK = wkT + (size_t)h * Dc * Cc;
    const _Float16* BV = wvT + (size_t)h * Dc * Cc;

    v8f aq[4], ak[4], av[4];
#pragma unroll
    for (int t = 0; t < 4; ++t) { aq[t] = zero8(); ak[t] = zero8(); av[t] = zero8(); }

    for (int k0 = 0; k0 < Cc; k0 += 32) {
        v16h a = load_frag(A + k0, Cc);
#pragma unroll
        for (int t = 0; t < 4; ++t) {
            v16h bq = load_frag(BQ + (size_t)(t * 16) * Cc + k0, Cc);
            aq[t] = wmma32(a, bq, aq[t]);
            v16h bk = load_frag(BK + (size_t)(t * 16) * Cc + k0, Cc);
            ak[t] = wmma32(a, bk, ak[t]);
            v16h bv = load_frag(BV + (size_t)(t * 16) * Cc + k0, Cc);
            av[t] = wmma32(a, bv, av[t]);
        }
    }

    const size_t qbase = (size_t)(b * Hc + h) * Tc + row0;
    const size_t vbase = (size_t)(b * Hc + h) * Dc;
#pragma unroll
    for (int t = 0; t < 4; ++t) {
#pragma unroll
        for (int i = 0; i < 8; ++i) {
            int m = i + half * 8;
            int d = t * 16 + ln;
            qh[(qbase + m) * Dc + d] = (_Float16)aq[t][i];
            kh[(qbase + m) * Dc + d] = (_Float16)ak[t][i];
            vT[(vbase + d) * Tc + row0 + m] = (_Float16)av[t][i];
        }
    }
}

// ---------------- flash attention ----------------
// grid: B*H*(T/64), 128 threads (4 waves); each wave owns 16 query rows.
// K/V 64x64 tiles are DMA'd into LDS once per workgroup by the Tensor Data
// Mover (double-buffered, overlapping the WMMA work of the previous block).
__global__ __launch_bounds__(128) void k_attn(
    const _Float16* __restrict__ qh,   // [B,H,T,D]
    const _Float16* __restrict__ kh,   // [B,H,T,D]
    const _Float16* __restrict__ vT,   // [B,H,D,T]
    _Float16* __restrict__ oh) {       // [B,T,H*D]
    __shared__ _Float16 Kt[2][64 * 64];   // rows = key s (64), cols = d (64)
    __shared__ _Float16 Vt[2][64 * 64];   // rows = d (64),     cols = key s (64)
    __shared__ _Float16 Pl[4][16 * 64];   // per-wave P staging

    const int ntb  = Tc / 64;
    const int tb   = blockIdx.x % ntb;
    const int bh   = blockIdx.x / ntb;
    const int h    = bh % Hc;
    const int b    = bh / Hc;
    const int wave = threadIdx.x >> 5;
    const int lane = threadIdx.x & 31;
    const int ln   = lane & 15;
    const int half = lane >> 4;
    const int q0   = tb * 64 + wave * 16;           // wave's first query row
    const float scale = 0.03125f;                   // 1/sqrt(C) = 1/32
    const float NEG_INF = -__builtin_inff();

    const _Float16* Qb = qh + ((size_t)(b * Hc + h) * Tc + q0) * Dc;
    const _Float16* Kb = kh + (size_t)(b * Hc + h) * Tc * Dc;
    const _Float16* Vb = vT + (size_t)(b * Hc + h) * Dc * Tc;

    v16h qf0 = load_frag(Qb, Dc);
    v16h qf1 = load_frag(Qb + 32, Dc);

    v8f O[4];
#pragma unroll
    for (int t = 0; t < 4; ++t) O[t] = zero8();
    v8f mrow, lrow;
#pragma unroll
    for (int i = 0; i < 8; ++i) { mrow[i] = NEG_INF; lrow[i] = 0.f; }

#if HAS_TDM
    if (wave == 0) {
        tdm_load_tile_f16(Kb, (unsigned int)(uintptr_t)&Kt[0][0], 64, 64, 64);
        tdm_load_tile_f16(Vb, (unsigned int)(uintptr_t)&Vt[0][0], 64, 64, (unsigned long long)Tc);
    }
#endif

    // uniform loop: every wave's diagonal block is s0 = tb*64
    for (int blk = 0; blk <= tb; ++blk) {
        const int s0 = blk * 64;
#if HAS_TDM
        const int buf = blk & 1;
        if (wave == 0) __builtin_amdgcn_s_wait_tensorcnt(0);
        __syncthreads();   // publish tile(blk); all waves done with buf^1
        if (wave == 0 && blk < tb) {
            const int s1 = s0 + 64;
            tdm_load_tile_f16(Kb + (size_t)s1 * Dc,
                              (unsigned int)(uintptr_t)&Kt[buf ^ 1][0], 64, 64, 64);
            tdm_load_tile_f16(Vb + s1,
                              (unsigned int)(uintptr_t)&Vt[buf ^ 1][0], 64, 64,
                              (unsigned long long)Tc);
        }
#else
        const int buf = 0;
        __syncthreads();
        {   // cooperative staging fallback
            const _Float16* src = Kb + (size_t)s0 * Dc;      // contiguous 8KB
            v8h* dstk = (v8h*)&Kt[0][0];
            for (int c = threadIdx.x; c < 512; c += 128)
                dstk[c] = ((const v8h*)src)[c];
            const _Float16* vs = Vb + s0;
            v8h* dstv = (v8h*)&Vt[0][0];
            for (int c = threadIdx.x; c < 512; c += 128) {
                int row = c >> 3, off = (c & 7) * 8;
                dstv[c] = *(const v8h*)(vs + (size_t)row * Tc + off);
            }
        }
        __syncthreads();
#endif
        const _Float16* Kl = &Kt[buf][0];
        const _Float16* Vl = &Vt[buf][0];

        // ---- S = Q Kᵀ (16x64 per wave), B-fragments from LDS ----
        v8f S[4];
#pragma unroll
        for (int t = 0; t < 4; ++t) {
            const _Float16* kp = Kl + (t * 16) * 64;
            v16h b0 = load_frag(kp, 64);
            v16h b1 = load_frag(kp + 32, 64);
            v8f s = zero8();
            s = wmma32(qf0, b0, s);
            s = wmma32(qf1, b1, s);
            S[t] = s;
        }
        // ---- scale + causal mask (only the last block is partial) ----
        const bool edge = (s0 + 63 > q0);
#pragma unroll
        for (int t = 0; t < 4; ++t) {
#pragma unroll
            for (int i = 0; i < 8; ++i) {
                float v = S[t][i] * scale;
                if (edge) {
                    int sg = s0 + t * 16 + ln;
                    int qg = q0 + i + half * 8;
                    v = (sg > qg) ? NEG_INF : v;
                }
                S[t][i] = v;
            }
        }
        // ---- online softmax stats ----
        v8f bm = vmax8(vmax8(S[0], S[1]), vmax8(S[2], S[3]));
        bm = reduce16_max(bm);
        v8f mnew = vmax8(mrow, bm);
        v8f fac;
#pragma unroll
        for (int i = 0; i < 8; ++i) fac[i] = __expf(mrow[i] - mnew[i]);
#pragma unroll
        for (int t = 0; t < 4; ++t)
#pragma unroll
            for (int i = 0; i < 8; ++i) O[t][i] *= fac[i];
        v8f rs = zero8();
#pragma unroll
        for (int t = 0; t < 4; ++t) {
#pragma unroll
            for (int i = 0; i < 8; ++i) {
                float p = __expf(S[t][i] - mnew[i]);
                S[t][i] = p;
                rs[i] += p;
            }
        }
        rs = reduce16_sum(rs);
#pragma unroll
        for (int i = 0; i < 8; ++i) lrow[i] = lrow[i] * fac[i] + rs[i];
        mrow = mnew;

        // ---- stage P (16x64 f16) in wave-private LDS, reload as A-fragments ----
        _Float16* Pw = &Pl[wave][0];
#pragma unroll
        for (int t = 0; t < 4; ++t)
#pragma unroll
            for (int i = 0; i < 8; ++i)
                Pw[(i + half * 8) * 64 + t * 16 + ln] = (_Float16)S[t][i];
        asm volatile("s_wait_dscnt 0" ::: "memory");
        v16h p0 = load_frag(Pw, 64);
        v16h p1 = load_frag(Pw + 32, 64);

        // ---- O += P V, B-fragments from LDS ----
#pragma unroll
        for (int t = 0; t < 4; ++t) {
            const _Float16* vp = Vl + (t * 16) * 64;
            v16h v0 = load_frag(vp, 64);
            v16h v1 = load_frag(vp + 32, 64);
            O[t] = wmma32(p0, v0, O[t]);
            O[t] = wmma32(p1, v1, O[t]);
        }
    }

    // ---- normalize, store [B,T,H*D] f16 ----
    v8f inv;
#pragma unroll
    for (int i = 0; i < 8; ++i) inv[i] = 1.f / lrow[i];
    const size_t ob = (size_t)b * Tc + q0;
#pragma unroll
    for (int t = 0; t < 4; ++t)
#pragma unroll
        for (int i = 0; i < 8; ++i)
            oh[(ob + i + half * 8) * KD + h * Dc + t * 16 + ln] =
                (_Float16)(O[t][i] * inv[i]);
}

// ---------------- output projection: out = oh @ Wo + bo ----------------
// grid: (B*T/64)*(C/64), 256 threads (8 waves); wave -> 16x32 slice
__global__ __launch_bounds__(256) void k_out(
    const _Float16* __restrict__ oh,    // [B*T, KD]
    const _Float16* __restrict__ woT,   // [C, KD]
    const float* __restrict__ bo,       // [C]
    float* __restrict__ out) {          // [B*T, C]
    const int nct = Cc / 64;
    const int ct  = blockIdx.x % nct;
    const int rt  = blockIdx.x / nct;
    const int wave = threadIdx.x >> 5;
    const int lane = threadIdx.x & 31;
    const int ln   = lane & 15;
    const int half = lane >> 4;
    const int r0 = rt * 64 + (wave >> 1) * 16;
    const int c0 = ct * 64 + (wave & 1) * 32;

    const _Float16* A = oh + (size_t)r0 * KD;
    v8f acc[2];
    acc[0] = zero8(); acc[1] = zero8();
    for (int k0 = 0; k0 < KD; k0 += 32) {
        v16h a = load_frag(A + k0, KD);
#pragma unroll
        for (int t = 0; t < 2; ++t) {
            v16h bb = load_frag(woT + (size_t)(c0 + t * 16) * KD + k0, KD);
            acc[t] = wmma32(a, bb, acc[t]);
        }
    }
#pragma unroll
    for (int t = 0; t < 2; ++t) {
#pragma unroll
        for (int i = 0; i < 8; ++i) {
            int col = c0 + t * 16 + ln;
            int row = r0 + i + half * 8;
            out[(size_t)row * Cc + col] = acc[t][i] + bo[col];
        }
    }
}

// ---------------- host launch ----------------
extern "C" void kernel_launch(void* const* d_in, const int* in_sizes, int n_in,
                              void* d_out, int out_size, void* d_ws, size_t ws_size,
                              hipStream_t stream) {
    const float* x  = (const float*)d_in[0];
    const float* Wq = (const float*)d_in[1];
    const float* Wk = (const float*)d_in[2];
    const float* Wv = (const float*)d_in[3];
    const float* Wo = (const float*)d_in[4];
    const float* bo = (const float*)d_in[5];
    float* out = (float*)d_out;

    char* ws = (char*)d_ws;
    size_t off = 0;
    auto carve = [&](size_t elems) {
        _Float16* p = (_Float16*)(ws + off);
        off += elems * sizeof(_Float16);
        off = (off + 255) & ~(size_t)255;
        return p;
    };
    _Float16* xh  = carve((size_t)Bc * Tc * Cc);
    _Float16* wqT = carve((size_t)Hc * Dc * Cc);
    _Float16* wkT = carve((size_t)Hc * Dc * Cc);
    _Float16* wvT = carve((size_t)Hc * Dc * Cc);
    _Float16* woT = carve((size_t)Cc * KD);
    _Float16* qh  = carve((size_t)Bc * Hc * Tc * Dc);
    _Float16* kh  = carve((size_t)Bc * Hc * Tc * Dc);
    _Float16* vT  = carve((size_t)Bc * Hc * Dc * Tc);
    _Float16* ohb = carve((size_t)Bc * Tc * KD);
    (void)ws_size; (void)n_in; (void)in_sizes; (void)out_size;

    int n;
    n = Bc * Tc * Cc;
    k_cvt<<<(n + 255) / 256, 256, 0, stream>>>(x, xh, n);
    n = Hc * Dc * Cc;
    k_cvt_tr_w<<<(n + 255) / 256, 256, 0, stream>>>(Wq, wqT);
    k_cvt_tr_w<<<(n + 255) / 256, 256, 0, stream>>>(Wk, wkT);
    k_cvt_tr_w<<<(n + 255) / 256, 256, 0, stream>>>(Wv, wvT);
    n = Cc * KD;
    k_cvt_tr_wo<<<(n + 255) / 256, 256, 0, stream>>>(Wo, woT);

    const int nblk = Bc * Hc * (Tc / 64);
    k_qkv<<<nblk, 128, 0, stream>>>(xh, wqT, wkT, wvT, qh, kh, vT);
    k_attn<<<nblk, 128, 0, stream>>>(qh, kh, vT, ohb);

    const int oblk = (Bc * Tc / 64) * (Cc / 64);
    k_out<<<oblk, 256, 0, stream>>>(ohb, woT, bo, out);
}
